// FactorizedTransformerEncoderLayer_13975823581384
// MI455X (gfx1250) — matrix operand order repaired and checked by reading, not compiled, not run on hardware
//
#include <hip/hip_runtime.h>
#include <hip/hip_bf16.h>

// ---------------------------------------------------------------------------
// Problem constants (from the reference)
// ---------------------------------------------------------------------------
static constexpr int Bc  = 8;
static constexpr int Sc  = 1024;
static constexpr int Dc  = 1024;
static constexpr int Hc  = 16;
static constexpr int HIDc = 4096;
static constexpr int Rc  = 256;
static constexpr int DHc = 64;
static constexpr int Mrows = Bc * Sc;     // 8192 tokens
static constexpr float EPSc = 1e-5f;

// ---------------------------------------------------------------------------
// Types for WMMA (CDNA5 / gfx1250, wave32)
// ---------------------------------------------------------------------------
typedef __bf16 bf16;
typedef __bf16  bf16x16 __attribute__((ext_vector_type(16)));
typedef __bf16  bf16x8  __attribute__((ext_vector_type(8)));
typedef float   f32x8   __attribute__((ext_vector_type(8)));

#define WMMA_BF16(a, b, c) \
  __builtin_amdgcn_wmma_f32_16x16x32_bf16(false, (a), false, (b), (short)0, (c), false, false)

#define CONCAT16(lo, up) \
  __builtin_shufflevector((lo), (up), 0, 1, 2, 3, 4, 5, 6, 7, 8, 9, 10, 11, 12, 13, 14, 15)

// ---------------------------------------------------------------------------
// Tensor Data Mover availability (CDNA5). Guarded so either toolchain builds.
// ---------------------------------------------------------------------------
#if defined(__AMDGCN__) && __has_builtin(__builtin_amdgcn_tensor_load_to_lds) && \
    __has_builtin(__builtin_amdgcn_s_wait_tensorcnt)
#define USE_TDM 1
#else
#define USE_TDM 0
#endif

static constexpr int BM = 128;
static constexpr int BN = 64;
static constexpr int BK = 32;
static constexpr int LDSP = 40;   // padded LDS K-pitch: 64B data + 16B pad = 80B

#if USE_TDM
typedef unsigned u32x4 __attribute__((ext_vector_type(4)));
typedef int      i32x4 __attribute__((ext_vector_type(4)));
typedef int      i32x8 __attribute__((ext_vector_type(8)));

// Generic LDS pointer -> block-relative LDS byte offset (low 32 bits).
__device__ __forceinline__ unsigned lds_byte_off(const void* p) {
  return (unsigned)(uintptr_t)p;
}

// Issue one TDM DMA: rows x BK bf16 tile, global row pitch = kstride elems,
// into LDS with 80-byte row pitch (16-DWORD pad interval + 4-DWORD pad).
__device__ __forceinline__ void tdm_load_tile(const bf16* gsrc, unsigned lds_off,
                                              int rows, int kstride_elems) {
  const unsigned long long ga = (unsigned long long)(uintptr_t)gsrc;
  u32x4 g0;
  g0[0] = 1u;                                                  // count=1, user D#
  g0[1] = lds_off;                                             // lds_addr
  g0[2] = (unsigned)ga;                                        // global_addr[31:0]
  g0[3] = (unsigned)((ga >> 32) & 0x01FFFFFFull) | (2u << 30); // addr[56:32] | type=2
  i32x8 g1;
  // data_size=2B, pad_enable, pad_interval=16 DW (code 3), pad_amount=4 DW (code 3)
  g1[0] = (1 << 16) | (1 << 20) | (3 << 22) | (3 << 25);
  g1[1] = (int)(((unsigned)kstride_elems & 0xFFFFu) << 16);    // tensor_dim0[15:0]
  g1[2] = (int)(((unsigned)kstride_elems >> 16) & 0xFFFFu);    // dim0[31:16] | dim1[15:0]=0
  g1[3] = (int)(16u | ((unsigned)BK << 16));                   // dim1[31:16]=16 (dim1=1M) | tile_dim0=32
  g1[4] = rows;                                                // tile_dim1=rows | tile_dim2=0
  g1[5] = kstride_elems;                                       // tensor_dim0_stride[31:0]
  g1[6] = 0;                                                   // stride0 hi | stride1 lo
  g1[7] = 0;                                                   // stride1 hi
  i32x4 gz = {0, 0, 0, 0};
#if __clang_major__ >= 23
  i32x8 gz8 = {0, 0, 0, 0, 0, 0, 0, 0};
  __builtin_amdgcn_tensor_load_to_lds(g0, g1, gz, gz, gz8, 0);
#else
  __builtin_amdgcn_tensor_load_to_lds(g0, g1, gz, gz, 0);
#endif
}
#endif  // USE_TDM

// ---------------------------------------------------------------------------
// fp32 -> bf16 conversion (elementwise)
// ---------------------------------------------------------------------------
__global__ __launch_bounds__(256) void cvt_f32_bf16(const float* __restrict__ in,
                                                    bf16* __restrict__ out, int n) {
  int i = blockIdx.x * 256 + threadIdx.x;
  if (i < n) out[i] = (bf16)in[i];
}

// ---------------------------------------------------------------------------
// Tiled WMMA GEMM:  C[M,N] = A[M,K] * W[N,K]^T (+bias) (+exact GELU)
// Block tile 128(M) x 64(N), BK=32, 256 threads = 8 waves, wave tile 32x32.
// TDM double-buffered LDS staging when available.
// ---------------------------------------------------------------------------
template <bool OUT_BF16, bool DO_GELU>
__global__ __launch_bounds__(256) void gemm_wmma(const bf16* __restrict__ A,
                                                 const bf16* __restrict__ W,
                                                 const float* __restrict__ bias,
                                                 void* __restrict__ Cout,
                                                 int M, int N, int K) {
#if USE_TDM
  __shared__ __align__(16) bf16 As[2][BM][LDSP];
  __shared__ __align__(16) bf16 Bs[2][BN][LDSP];
#else
  __shared__ __align__(16) bf16 As[1][BM][LDSP];
  __shared__ __align__(16) bf16 Bs[1][BN][LDSP];
#endif

  const int tid  = threadIdx.x;
  const int lane = tid & 31;
  const int wave = tid >> 5;
  const int wm   = (wave & 3) * 32;   // 4 waves along M
  const int wn   = (wave >> 2) * 32;  // 2 waves along N
  const int bm   = blockIdx.y * BM;
  const int bn   = blockIdx.x * BN;

  const int r2 = lane & 15;
  const int hi = lane >> 4;

  f32x8 acc[2][2] = {};
  int p = 0;

#if USE_TDM
  // Make the DMA-written LDS visibly "stored-to" (and bounded) for the compiler.
  for (int i = tid; i < (int)(sizeof(As) / sizeof(int)); i += 256) ((int*)As)[i] = 0;
  for (int i = tid; i < (int)(sizeof(Bs) / sizeof(int)); i += 256) ((int*)Bs)[i] = 0;
  __syncthreads();
  if (wave == 0) {
    tdm_load_tile(A + (size_t)bm * K, lds_byte_off(&As[0][0][0]), BM, K);
    tdm_load_tile(W + (size_t)bn * K, lds_byte_off(&Bs[0][0][0]), BN, K);
  }
#else
  const int la_r = tid >> 2;          // 0..63
  const int la_c = (tid & 3) * 8;     // 0,8,16,24
#endif

  for (int k0 = 0; k0 < K; k0 += BK) {
#if USE_TDM
    const bool more = (k0 + BK) < K;
    if (wave == 0) {
      if (more) {
        tdm_load_tile(A + (size_t)bm * K + (k0 + BK), lds_byte_off(&As[p ^ 1][0][0]), BM, K);
        tdm_load_tile(W + (size_t)bn * K + (k0 + BK), lds_byte_off(&Bs[p ^ 1][0][0]), BN, K);
      }
      if (more) __builtin_amdgcn_s_wait_tensorcnt(2);  // oldest pair (buffer p) done
      else      __builtin_amdgcn_s_wait_tensorcnt(0);
    }
    __syncthreads();
#else
    const bf16* ga0 = A + (size_t)(bm + la_r) * K + (k0 + la_c);
    const bf16* ga1 = A + (size_t)(bm + la_r + 64) * K + (k0 + la_c);
    const bf16* gb  = W + (size_t)(bn + la_r) * K + (k0 + la_c);
    *(bf16x8*)&As[0][la_r][la_c]      = *(const bf16x8*)ga0;
    *(bf16x8*)&As[0][la_r + 64][la_c] = *(const bf16x8*)ga1;
    *(bf16x8*)&Bs[0][la_r][la_c]      = *(const bf16x8*)gb;
    if (k0 + BK < K) {
      __builtin_prefetch(ga0 + BK, 0, 1);
      __builtin_prefetch(ga1 + BK, 0, 1);
      __builtin_prefetch(gb + BK, 0, 1);
    }
    __syncthreads();
#endif

    // ---- fragments (ISA bf16 A 16x32 / B 32x16 layouts) ----
    bf16x16 afrag[2], bfrag[2];
#pragma unroll
    for (int i = 0; i < 2; ++i) {
      const int row = wm + i * 16 + r2;
      bf16x8 lo = *(const bf16x8*)&As[p][row][hi * 8];
      bf16x8 up = *(const bf16x8*)&As[p][row][16 + hi * 8];
      afrag[i] = CONCAT16(lo, up);
    }
#pragma unroll
    for (int j = 0; j < 2; ++j) {
      const int row = wn + j * 16 + r2;
      bf16x8 lo = *(const bf16x8*)&Bs[p][row][hi * 8];
      bf16x8 up = *(const bf16x8*)&Bs[p][row][16 + hi * 8];
      bfrag[j] = CONCAT16(lo, up);
    }

#pragma unroll
    for (int i = 0; i < 2; ++i)
#pragma unroll
      for (int j = 0; j < 2; ++j)
        acc[i][j] = WMMA_BF16(afrag[i], bfrag[j], acc[i][j]);
    __syncthreads();
#if USE_TDM
    p ^= 1;
#endif
  }

  // ---- epilogue: C layout => VGPR q: row = q + hi*8, col = lane&15 ----
#pragma unroll
  for (int i = 0; i < 2; ++i) {
#pragma unroll
    for (int j = 0; j < 2; ++j) {
      const int col = bn + wn + j * 16 + r2;
      const float bv = bias ? bias[col] : 0.0f;
#pragma unroll
      for (int q = 0; q < 8; ++q) {
        const int row = bm + wm + i * 16 + q + hi * 8;
        float v = acc[i][j][q] + bv;
        if (DO_GELU) v = 0.5f * v * (1.0f + erff(v * 0.70710678118f));
        if (OUT_BF16) ((bf16*)Cout)[(size_t)row * N + col] = (bf16)v;
        else          ((float*)Cout)[(size_t)row * N + col] = v;
      }
    }
  }
}

// ---------------------------------------------------------------------------
// Causal flash attention, one wave per 16 query rows, WMMA for QK^T and PV.
// Q,K,V,O are bf16 in [B, S, D] with head h occupying columns h*64..h*64+63.
// ---------------------------------------------------------------------------
__global__ __launch_bounds__(256) void attn_flash(const bf16* __restrict__ Q,
                                                  const bf16* __restrict__ Kb,
                                                  const bf16* __restrict__ V,
                                                  bf16* __restrict__ O) {
  __shared__ __align__(16) bf16 Plds[8][16][LDSP];  // wave-private P tiles

  const int tid  = threadIdx.x;
  const int lane = tid & 31;
  const int wave = tid >> 5;
  const int r2 = lane & 15;
  const int hi = lane >> 4;

  const int bh = blockIdx.y;
  const int b  = bh >> 4;          // H = 16
  const int h  = bh & 15;
  const int qbase = (blockIdx.x * 8 + wave) * 16;

  const size_t base = (size_t)b * Sc * Dc + (size_t)h * DHc;  // + tok*Dc + dh

  // ---- Q fragments: 16 rows x 64 dh -> two A-frags (K=32 each) ----
  bf16x16 qf[2];
  {
    const bf16* qr = Q + base + (size_t)(qbase + r2) * Dc;
#pragma unroll
    for (int c = 0; c < 2; ++c) {
      bf16x8 lo = *(const bf16x8*)(qr + c * 32 + hi * 8);
      bf16x8 up = *(const bf16x8*)(qr + c * 32 + 16 + hi * 8);
      qf[c] = CONCAT16(lo, up);
    }
  }

  float rmax[8], rsum[8];
#pragma unroll
  for (int q = 0; q < 8; ++q) { rmax[q] = -3.0e38f; rsum[q] = 0.0f; }
  f32x8 ctx[4] = {};

  for (int k0 = 0; k0 < qbase + 16; k0 += 32) {
    // ---- scores S = (Q K^T) * 0.125 for 32 keys: two 16x16 accums ----
    f32x8 sacc[2] = {};
#pragma unroll
    for (int g = 0; g < 2; ++g) {
      const bf16* kr = Kb + base + (size_t)(k0 + g * 16 + r2) * Dc;
#pragma unroll
      for (int c = 0; c < 2; ++c) {
        bf16x8 lo = *(const bf16x8*)(kr + c * 32 + hi * 8);
        bf16x8 up = *(const bf16x8*)(kr + c * 32 + 16 + hi * 8);
        bf16x16 kf = CONCAT16(lo, up);
        sacc[g] = WMMA_BF16(qf[c], kf, sacc[g]);
      }
    }

    // ---- causal mask + online softmax (rows live in 16-lane halves) ----
#pragma unroll
    for (int q = 0; q < 8; ++q) {
      const int row = qbase + q + hi * 8;
      float m = rmax[q];
#pragma unroll
      for (int g = 0; g < 2; ++g) {
        const int key = k0 + g * 16 + r2;
        float sv = sacc[g][q] * 0.125f;
        sv = (key <= row) ? sv : -3.0e38f;
        sacc[g][q] = sv;
        m = fmaxf(m, sv);
      }
#pragma unroll
      for (int off = 8; off >= 1; off >>= 1) m = fmaxf(m, __shfl_xor(m, off, 32));
      const float corr = __expf(rmax[q] - m);
      float psum = 0.0f;
#pragma unroll
      for (int g = 0; g < 2; ++g) {
        float e = __expf(sacc[g][q] - m);
        sacc[g][q] = e;
        psum += e;
      }
#pragma unroll
      for (int off = 8; off >= 1; off >>= 1) psum += __shfl_xor(psum, off, 32);
      rsum[q] = rsum[q] * corr + psum;
      rmax[q] = m;
#pragma unroll
      for (int c2 = 0; c2 < 4; ++c2) ctx[c2][q] *= corr;
    }

    // ---- P: C-layout f32 -> A-layout bf16 via wave-private LDS ----
#pragma unroll
    for (int g = 0; g < 2; ++g)
#pragma unroll
      for (int q = 0; q < 8; ++q)
        Plds[wave][q + hi * 8][g * 16 + r2] = (bf16)sacc[g][q];
    bf16x16 pf;
    {
      bf16x8 lo = *(const bf16x8*)&Plds[wave][r2][hi * 8];
      bf16x8 up = *(const bf16x8*)&Plds[wave][r2][16 + hi * 8];
      pf = CONCAT16(lo, up);
    }

    // ---- ctx += P * V : V B-frags are columns of V (strided gathers) ----
#pragma unroll
    for (int c2 = 0; c2 < 4; ++c2) {
      bf16x16 vf;
      const bf16* vb = V + base + (size_t)c2 * 16 + r2;  // fixed dh column
#pragma unroll
      for (int j = 0; j < 8; ++j) {
        const int key = k0 + hi * 8 + 2 * j + (j >= 4 ? 8 : 0);
        vf[2 * j]     = vb[(size_t)key * Dc];
        vf[2 * j + 1] = vb[(size_t)(key + 1) * Dc];
      }
      ctx[c2] = WMMA_BF16(pf, vf, ctx[c2]);
    }
  }

  // ---- normalize and store ----
#pragma unroll
  for (int c2 = 0; c2 < 4; ++c2)
#pragma unroll
    for (int q = 0; q < 8; ++q) {
      const int tok = qbase + q + hi * 8;
      const float v = ctx[c2][q] / rsum[q];
      O[base + (size_t)tok * Dc + c2 * 16 + r2] = (bf16)v;
    }
}

// ---------------------------------------------------------------------------
// Fused residual + LayerNorm over D=1024; one 256-thread block per row.
// ---------------------------------------------------------------------------
__global__ __launch_bounds__(256) void resid_layernorm(const float* __restrict__ a,
                                                       const float* __restrict__ b,
                                                       const float* __restrict__ g,
                                                       const float* __restrict__ beta,
                                                       float* __restrict__ out_f32,
                                                       bf16* __restrict__ out_bf16) {
  __shared__ float red[8];
  const int row = blockIdx.x;
  const int tid = threadIdx.x;
  const float* pa = a + (size_t)row * Dc;
  const float* pb = b + (size_t)row * Dc;

  float v[4];
  float s = 0.0f;
#pragma unroll
  for (int i = 0; i < 4; ++i) {
    const int c = tid + i * 256;
    v[i] = pa[c] + pb[c];
    s += v[i];
  }
#pragma unroll
  for (int off = 16; off >= 1; off >>= 1) s += __shfl_xor(s, off, 32);
  if ((tid & 31) == 0) red[tid >> 5] = s;
  __syncthreads();
  s = 0.0f;
#pragma unroll
  for (int w = 0; w < 8; ++w) s += red[w];
  const float mu = s * (1.0f / Dc);
  __syncthreads();

  float sq = 0.0f;
#pragma unroll
  for (int i = 0; i < 4; ++i) {
    const float d = v[i] - mu;
    sq += d * d;
  }
#pragma unroll
  for (int off = 16; off >= 1; off >>= 1) sq += __shfl_xor(sq, off, 32);
  if ((tid & 31) == 0) red[tid >> 5] = sq;
  __syncthreads();
  sq = 0.0f;
#pragma unroll
  for (int w = 0; w < 8; ++w) sq += red[w];
  const float rstd = rsqrtf(sq * (1.0f / Dc) + EPSc);

#pragma unroll
  for (int i = 0; i < 4; ++i) {
    const int c = tid + i * 256;
    const float o = (v[i] - mu) * rstd * g[c] + beta[c];
    out_f32[(size_t)row * Dc + c] = o;
    if (out_bf16) out_bf16[(size_t)row * Dc + c] = (bf16)o;
  }
}

// ---------------------------------------------------------------------------
// Host-side orchestration
// ---------------------------------------------------------------------------
static inline size_t align256(size_t x) { return (x + 255) & ~(size_t)255; }

extern "C" void kernel_launch(void* const* d_in, const int* in_sizes, int n_in,
                              void* d_out, int out_size, void* d_ws, size_t ws_size,
                              hipStream_t stream) {
  (void)in_sizes; (void)n_in; (void)out_size; (void)ws_size;

  const float* src = (const float*)d_in[0];
  const float* Wq  = (const float*)d_in[1];
  const float* bq  = (const float*)d_in[2];
  const float* Wk  = (const float*)d_in[3];
  const float* bk  = (const float*)d_in[4];
  const float* Wv  = (const float*)d_in[5];
  const float* bv  = (const float*)d_in[6];
  const float* Wo  = (const float*)d_in[7];
  const float* bo  = (const float*)d_in[8];
  const float* g1  = (const float*)d_in[9];
  const float* b1  = (const float*)d_in[10];
  const float* W1a = (const float*)d_in[11];
  const float* W1b = (const float*)d_in[12];
  const float* W2a = (const float*)d_in[13];
  const float* W2b = (const float*)d_in[14];
  const float* g2  = (const float*)d_in[15];
  const float* b2  = (const float*)d_in[16];
  float* out = (float*)d_out;

  // ---- workspace carve-up ----
  char* ws = (char*)d_ws;
  size_t off = 0;
  auto take = [&](size_t bytes) { char* p = ws + off; off += align256(bytes); return p; };

  bf16* sA   = (bf16*)take((size_t)Mrows * Dc * 2);
  bf16* wq   = (bf16*)take((size_t)Dc * Dc * 2);
  bf16* wk   = (bf16*)take((size_t)Dc * Dc * 2);
  bf16* wv   = (bf16*)take((size_t)Dc * Dc * 2);
  bf16* wo   = (bf16*)take((size_t)Dc * Dc * 2);
  bf16* w1a  = (bf16*)take((size_t)Rc * Dc * 2);
  bf16* w1b  = (bf16*)take((size_t)HIDc * Rc * 2);
  bf16* w2a  = (bf16*)take((size_t)Rc * HIDc * 2);
  bf16* w2b  = (bf16*)take((size_t)Dc * Rc * 2);
  bf16* Qb   = (bf16*)take((size_t)Mrows * Dc * 2);
  bf16* Kbuf = (bf16*)take((size_t)Mrows * Dc * 2);
  bf16* Vb   = (bf16*)take((size_t)Mrows * Dc * 2);
  bf16* ctxb = (bf16*)take((size_t)Mrows * Dc * 2);
  float* attn_f = (float*)take((size_t)Mrows * Dc * 4);
  float* x_f    = (float*)take((size_t)Mrows * Dc * 4);
  bf16*  x_bf   = (bf16*)take((size_t)Mrows * Dc * 2);
  bf16*  t1     = (bf16*)take((size_t)Mrows * Rc * 2);
  bf16*  hbuf   = (bf16*)take((size_t)Mrows * HIDc * 2);
  bf16*  t2     = (bf16*)take((size_t)Mrows * Rc * 2);
  float* ffn    = (float*)take((size_t)Mrows * Dc * 4);

  auto cvt = [&](const float* in, bf16* o, int n) {
    cvt_f32_bf16<<<(n + 255) / 256, 256, 0, stream>>>(in, o, n);
  };

  // ---- 1) downconvert activations & weights to bf16 ----
  cvt(src, sA, Mrows * Dc);
  cvt(Wq, wq, Dc * Dc);
  cvt(Wk, wk, Dc * Dc);
  cvt(Wv, wv, Dc * Dc);
  cvt(Wo, wo, Dc * Dc);
  cvt(W1a, w1a, Rc * Dc);
  cvt(W1b, w1b, HIDc * Rc);
  cvt(W2a, w2a, Rc * HIDc);
  cvt(W2b, w2b, Dc * Rc);

  // ---- 2) Q/K/V projections (M=8192, N=1024, K=1024) ----
  {
    dim3 grid(Dc / BN, Mrows / BM);
    gemm_wmma<true, false><<<grid, 256, 0, stream>>>(sA, wq, bq, Qb, Mrows, Dc, Dc);
    gemm_wmma<true, false><<<grid, 256, 0, stream>>>(sA, wk, bk, Kbuf, Mrows, Dc, Dc);
    gemm_wmma<true, false><<<grid, 256, 0, stream>>>(sA, wv, bv, Vb, Mrows, Dc, Dc);
  }

  // ---- 3) causal flash attention ----
  {
    dim3 grid(Sc / 128, Bc * Hc);  // 8 waves x 16 queries = 128 queries/block
    attn_flash<<<grid, 256, 0, stream>>>(Qb, Kbuf, Vb, ctxb);
  }

  // ---- 4) output projection (f32 out) ----
  {
    dim3 grid(Dc / BN, Mrows / BM);
    gemm_wmma<false, false><<<grid, 256, 0, stream>>>(ctxb, wo, bo, attn_f, Mrows, Dc, Dc);
  }

  // ---- 5) x = LN(src + attn_out) ----
  resid_layernorm<<<Mrows, 256, 0, stream>>>(src, attn_f, g1, b1, x_f, x_bf);

  // ---- 6) factorized FFN ----
  {
    dim3 gt1(Rc / BN, Mrows / BM);
    gemm_wmma<true, false><<<gt1, 256, 0, stream>>>(x_bf, w1a, nullptr, t1, Mrows, Rc, Dc);
    dim3 gh(HIDc / BN, Mrows / BM);
    gemm_wmma<true, true><<<gh, 256, 0, stream>>>(t1, w1b, nullptr, hbuf, Mrows, HIDc, Rc);
    dim3 gt2(Rc / BN, Mrows / BM);
    gemm_wmma<true, false><<<gt2, 256, 0, stream>>>(hbuf, w2a, nullptr, t2, Mrows, Rc, HIDc);
    dim3 gf(Dc / BN, Mrows / BM);
    gemm_wmma<false, false><<<gf, 256, 0, stream>>>(t2, w2b, nullptr, ffn, Mrows, Dc, Rc);
  }

  // ---- 7) out = LN(x + ffn) ----
  resid_layernorm<<<Mrows, 256, 0, stream>>>(x_f, ffn, g2, b2, out, nullptr);
}
